// RGCNBlockLayer_18657337933973
// MI455X (gfx1250) — matrix operand order repaired and checked by reading, not compiled, not run on hardware
//
#include <hip/hip_runtime.h>

#define IN_F 128
#define OUT_F 128
#define NB 8
#define SI 16
#define SO 16

typedef __bf16 bf16;
typedef bf16 v16bf __attribute__((ext_vector_type(16)));
typedef bf16 bf16x8 __attribute__((ext_vector_type(8)));
typedef float v8f   __attribute__((ext_vector_type(8)));
typedef unsigned int v4u __attribute__((ext_vector_type(4)));
typedef int v8i __attribute__((ext_vector_type(8)));
typedef int v4i __attribute__((ext_vector_type(4)));

union ABReg {
    v16bf v;
    struct Halves { bf16x8 lo; bf16x8 hi; } s;
};

__device__ __forceinline__ bf16x8 bzero8() {
    bf16x8 z;
#pragma unroll
    for (int i = 0; i < 8; ++i) z[i] = (bf16)0.0f;
    return z;
}

// ---------------- prep kernels ----------------

__global__ void k_init(float* __restrict__ agg, int* __restrict__ sorted,
                       int* __restrict__ counts, long aggN, int cap, int R) {
    long i = (long)blockIdx.x * blockDim.x + threadIdx.x;
    if (i < aggN) agg[i] = 0.0f;
    if (i < cap)  sorted[i] = -1;
    if (i < R)    counts[i] = 0;
}

__global__ void k_cvt_h(const float* __restrict__ h, bf16* __restrict__ hb, int n) {
    int i = blockIdx.x * blockDim.x + threadIdx.x;
    if (i < n) hb[i] = (bf16)h[i];
}

// wt[r][b][n][k] = weight[r][b*SI*SO + k*SO + n]  (transposed per block for WMMA B frags)
__global__ void k_cvt_w(const float* __restrict__ wsrc, bf16* __restrict__ wt, int total) {
    int i = blockIdx.x * blockDim.x + threadIdx.x;
    if (i >= total) return;
    int k = i & 15;
    int n = (i >> 4) & 15;
    int b = (i >> 8) & 7;
    int r = i >> 11;
    wt[i] = (bf16)wsrc[((r * NB + b) * SI + k) * SO + n];
}

// lwt[n][k] = loop_weight[k][n]
__global__ void k_cvt_lw(const float* __restrict__ lw, bf16* __restrict__ lwt, int total) {
    int i = blockIdx.x * blockDim.x + threadIdx.x;
    if (i >= total) return;
    int k = i & (IN_F - 1);
    int n = i >> 7;
    lwt[i] = (bf16)lw[k * OUT_F + n];
}

__global__ void k_hist(const int* __restrict__ etype, int* __restrict__ counts, int E) {
    int i = blockIdx.x * blockDim.x + threadIdx.x;
    if (i < E) atomicAdd(&counts[etype[i]], 1);
}

// bucket offsets padded to 16-edge tiles
__global__ void k_scan(const int* __restrict__ counts, int* __restrict__ wp, int R) {
    if (threadIdx.x == 0 && blockIdx.x == 0) {
        int off = 0;
        for (int r = 0; r < R; ++r) {
            wp[r] = off;
            off += ((counts[r] + 15) >> 4) << 4;
        }
    }
}

__global__ void k_scatter(const int* __restrict__ etype, int* __restrict__ wp,
                          int* __restrict__ sorted, int E) {
    int i = blockIdx.x * blockDim.x + threadIdx.x;
    if (i < E) {
        int p = atomicAdd(&wp[etype[i]], 1);
        sorted[p] = i;
    }
}

// ---------------- edge-tile block-diag GEMM (WMMA bf16, f32 acc) ----------------
// One wave per 16-edge tile; all real edges in a tile share one relation.
__global__ void __launch_bounds__(32) k_edge_gemm(
    const bf16* __restrict__ hb, const bf16* __restrict__ wt,
    const int* __restrict__ esrc, const int* __restrict__ edst,
    const int* __restrict__ etype, const int* __restrict__ sorted,
    float* __restrict__ agg) {
    __shared__ int s_src[16];
    __shared__ int s_dst[16];
    __shared__ int s_rel;
    __shared__ int s_any;

    const int lane = threadIdx.x;
    if (lane == 0) s_any = 0;
    __syncthreads();
    if (lane < 16) {
        int e = sorted[blockIdx.x * 16 + lane];
        int sv = -1, dv = -1;
        if (e >= 0) {
            sv = esrc[e];
            dv = edst[e];
            s_rel = etype[e];   // all valid lanes write the same value
            s_any = 1;
        }
        s_src[lane] = sv;
        s_dst[lane] = dv;
    }
    __syncthreads();
    if (s_any == 0) return;

    const int rel  = s_rel;
    const int col  = lane & 15;
    const int half = lane >> 4;
    const int src  = s_src[col];
    const bf16* wrel = wt + (size_t)rel * (NB * SI * SO);

    v8f c[NB];
#pragma unroll
    for (int b = 0; b < NB; ++b)
#pragma unroll
        for (int j = 0; j < 8; ++j) c[b][j] = 0.0f;

#pragma unroll
    for (int b = 0; b < NB; ++b) {
        ABReg a, bm;
        // A (16 edges x K32, K16..31 zero): lanes 0-15 hold K0-7, lanes 16-31 hold K8-15
        a.s.hi = bzero8();
        if (src >= 0)
            a.s.lo = *(const bf16x8*)(hb + (size_t)src * IN_F + b * SI + half * 8);
        else
            a.s.lo = bzero8();
        // B (K32 x 16, K16..31 zero): column col; lanes 0-15 hold K0-7, lanes 16-31 hold K8-15
        {
            const bf16* p = wrel + (b * SO + col) * SI + half * 8;
            bm.s.lo = *(const bf16x8*)p;
            bm.s.hi = bzero8();
        }
        c[b] = __builtin_amdgcn_wmma_f32_16x16x32_bf16(
            false, a.v, false, bm.v, (short)0, c[b], false, false);
    }

    // scatter-add: C VGPR j -> edge row (half*8 + j), output col (b*16 + col)
#pragma unroll
    for (int b = 0; b < NB; ++b) {
#pragma unroll
        for (int j = 0; j < 8; ++j) {
            int m = half * 8 + j;
            int d = s_dst[m];
            if (d >= 0)
                atomicAdd(&agg[(size_t)d * OUT_F + b * SO + col], c[b][j]);
        }
    }
}

// ---------------- fused self-loop GEMM + norm/bias/relu epilogue ----------------
// 128 threads = 4 waves; each wave owns a 16-node tile.
// loop_weight^T (32 KB) staged into LDS by the Tensor Data Mover (wave 0 issues
// one TDM descriptor; TENSORcnt waited before the block barrier).
__global__ void __launch_bounds__(128) k_final(
    const bf16* __restrict__ hb, const bf16* __restrict__ lwt,
    const float* __restrict__ agg, const float* __restrict__ norm,
    const float* __restrict__ bias, float* __restrict__ out, int N) {
    __shared__ bf16 s_lw[OUT_F * IN_F];   // 32 KB, [n][k], at LDS offset 0

    const int t = threadIdx.x;
    if (t < 32) {
        // TDM D#: 1-D tensor of 4096 x 8B elements (32 KB), tile = whole tensor.
        const unsigned long long ga = (unsigned long long)(uintptr_t)lwt;
        v4u g0;
        g0[0] = 1u;                                        // count=1, user descriptor
        g0[1] = 0u;                                        // lds_addr = 0 (s_lw)
        g0[2] = (unsigned)(ga & 0xFFFFFFFFu);              // global_addr[31:0]
        g0[3] = (unsigned)((ga >> 32) & 0x01FFFFFFu)       // global_addr[56:32]
              | (2u << 30);                                // type = 2 (image)
        v8i g1;
        g1[0] = (int)(3u << 16);                           // data_size = 8 bytes
        g1[1] = (int)((4096u & 0xFFFFu) << 16);            // tensor_dim0[15:0]
        g1[2] = (int)((4096u >> 16) | (1u << 16));         // tensor_dim0 hi | tensor_dim1=1
        g1[3] = (int)(4096u << 16);                        // tile_dim0 = 4096
        g1[4] = 0;                                         // tile_dim1/2 unused
        g1[5] = 4096;                                      // tensor_dim0_stride
        g1[6] = 0;
        g1[7] = 0;
        v4i z4 = {0, 0, 0, 0};
        v8i z8 = {0, 0, 0, 0, 0, 0, 0, 0};
        __builtin_amdgcn_tensor_load_to_lds(g0, g1, z4, z4, z8, 0);
        __builtin_amdgcn_s_wait_tensorcnt(0);
    }
    __syncthreads();

    const int wave = t >> 5;
    const int lane = t & 31;
    const int tileRow = (blockIdx.x * 4 + wave) * 16;
    if (tileRow >= N) return;

    const int col  = lane & 15;
    const int half = lane >> 4;
    const int am   = tileRow + col;

    ABReg a[4];
    if (am < N) {
#pragma unroll
        for (int kb = 0; kb < 4; ++kb) {
            const bf16* p = hb + (size_t)am * IN_F + kb * 32 + half * 8;
            a[kb].s.lo = *(const bf16x8*)p;        // K 0-7  / 8-15
            a[kb].s.hi = *(const bf16x8*)(p + 16); // K 16-23 / 24-31
        }
    } else {
#pragma unroll
        for (int kb = 0; kb < 4; ++kb) {
            a[kb].s.lo = bzero8();
            a[kb].s.hi = bzero8();
        }
    }

#pragma unroll
    for (int jb = 0; jb < 8; ++jb) {
        v8f c;
#pragma unroll
        for (int j = 0; j < 8; ++j) c[j] = 0.0f;
        const int n = jb * 16 + col;
#pragma unroll
        for (int kb = 0; kb < 4; ++kb) {
            ABReg bm;
            // B: lanes 0-15 K0-7,16-23, lanes 16-31 K8-15,24-31 of this k-block, column n
            const bf16* p = s_lw + n * IN_F + kb * 32 + half * 8;
            bm.s.lo = *(const bf16x8*)p;
            bm.s.hi = *(const bf16x8*)(p + 16);
            c = __builtin_amdgcn_wmma_f32_16x16x32_bf16(
                false, a[kb].v, false, bm.v, (short)0, c, false, false);
        }
        const float bn = bias[n];
#pragma unroll
        for (int j = 0; j < 8; ++j) {
            const int m = tileRow + half * 8 + j;
            if (m < N) {
                float v = agg[(size_t)m * OUT_F + n] * norm[m] + bn + c[j];
                out[(size_t)m * OUT_F + n] = v > 0.0f ? v : 0.0f;
            }
        }
    }
}

// ---------------- launcher ----------------

extern "C" void kernel_launch(void* const* d_in, const int* in_sizes, int n_in,
                              void* d_out, int out_size, void* d_ws, size_t ws_size,
                              hipStream_t stream) {
    const float* h      = (const float*)d_in[0];
    const float* norm   = (const float*)d_in[1];
    const int*   esrc   = (const int*)d_in[2];
    const int*   edst   = (const int*)d_in[3];
    const int*   etype  = (const int*)d_in[4];
    const float* weight = (const float*)d_in[5];
    const float* bias   = (const float*)d_in[6];
    const float* lw     = (const float*)d_in[7];
    float* out = (float*)d_out;

    const int N = in_sizes[1];
    const int E = in_sizes[2];
    const int R = in_sizes[5] / (NB * SI * SO);

    char* w = (char*)d_ws;
    size_t off = 0;
    auto align256 = [](size_t x) { return (x + 255) & ~(size_t)255; };

    float* agg = (float*)(w + off); off = align256(off + (size_t)N * OUT_F * sizeof(float));
    bf16*  hb  = (bf16*) (w + off); off = align256(off + (size_t)N * IN_F * sizeof(bf16));
    bf16*  wt  = (bf16*) (w + off); off = align256(off + (size_t)R * NB * SI * SO * sizeof(bf16));
    bf16*  lwt = (bf16*) (w + off); off = align256(off + (size_t)IN_F * OUT_F * sizeof(bf16));
    int* counts = (int*)(w + off);  off = align256(off + (size_t)R * sizeof(int));
    int* wp     = (int*)(w + off);  off = align256(off + (size_t)R * sizeof(int));
    const int cap = (((E + 15) / 16) + R) * 16;   // padded sorted-edge capacity
    int* sorted = (int*)(w + off);  off = align256(off + (size_t)cap * sizeof(int));
    (void)ws_size; (void)n_in; (void)out_size;

    const long aggN  = (long)N * OUT_F;
    const long initN = aggN > (long)cap ? aggN : (long)cap;

    k_init<<<dim3((unsigned)((initN + 255) / 256)), 256, 0, stream>>>(agg, sorted, counts, aggN, cap, R);
    k_cvt_h<<<dim3((unsigned)(((long)N * IN_F + 255) / 256)), 256, 0, stream>>>(h, hb, N * IN_F);
    k_cvt_w<<<dim3((unsigned)((R * NB * SI * SO + 255) / 256)), 256, 0, stream>>>(weight, wt, R * NB * SI * SO);
    k_cvt_lw<<<dim3((unsigned)((IN_F * OUT_F + 255) / 256)), 256, 0, stream>>>(lw, lwt, IN_F * OUT_F);
    k_hist<<<dim3((unsigned)((E + 255) / 256)), 256, 0, stream>>>(etype, counts, E);
    k_scan<<<1, 32, 0, stream>>>(counts, wp, R);
    k_scatter<<<dim3((unsigned)((E + 255) / 256)), 256, 0, stream>>>(etype, wp, sorted, E);
    k_edge_gemm<<<dim3((unsigned)(cap / 16)), 32, 0, stream>>>(hb, wt, esrc, edst, etype, sorted, agg);
    k_final<<<dim3((unsigned)((N + 63) / 64)), 128, 0, stream>>>(hb, lwt, agg, norm, bias, out, N);
}
